// RingDilatedAttentionHilbertOptimizedFixed_4784593568285
// MI455X (gfx1250) — matrix operand order repaired and checked by reading, not compiled, not run on hardware
//
#include <hip/hip_runtime.h>
#include <hip/hip_bf16.h>

// Per-position 16-head block-diagonal (2 groups x 8 heads) attention over d=64.
// Hilbert permutation + inverse cancel (position-wise computation), so this is
// a pure streaming kernel: ~256MB of HBM traffic -> ~11us floor at 23.3 TB/s.
// Matmuls use v_wmma_f32_16x16x32_f16 (f32 accumulate).

typedef __attribute__((ext_vector_type(16))) _Float16 v16h;
typedef __attribute__((ext_vector_type(8)))  float    v8f;
typedef __attribute__((ext_vector_type(4)))  float    f4;

static constexpr int   HEADS = 16;
static constexpr int   DIM   = 64;
static constexpr float SCALE = 0.125f; // 64^-0.5

__global__ __launch_bounds__(256, 1)
void hilbert_head_attn_kernel(const float* __restrict__ qg,
                              const float* __restrict__ kg,
                              const float* __restrict__ vg,
                              float* __restrict__ og,
                              int npos)
{
    const int lane = threadIdx.x & 31;
    const int wid  = threadIdx.x >> 5;
    const int pos  = blockIdx.x * 8 + wid;      // wave-uniform
    if (pos >= npos) return;                    // whole-wave exit, EXEC stays all-ones

    const int r  = lane & 15;                   // row/col index within 16
    const int hi = lane >> 4;                   // lane half
    const size_t base = (size_t)pos * (HEADS * DIM);

    const float* qrow = qg + base + (size_t)r * DIM;  // Q row m = r  (B operand)
    const float* krow = kg + base + (size_t)r * DIM;  // K row n = r  (A operand)

    // ---- Operands for S^T = K * Q^T, split into two d-chunks of 32 ----
    // A layout (16x32 f16): lane l holds M=l&15; halves 0..7 -> K=hi*8+0..7,
    // halves 8..15 -> K=hi*8+16..23.
    // B layout (32x16 f16): lane l holds N=l&15; halves 0..7 -> K=hi*8+0..7,
    // halves 8..15 -> K=hi*8+16..23.
    v16h aK0 = {}, aK1 = {}, bQ0 = {}, bQ1 = {};
#pragma unroll
    for (int i = 0; i < 8; ++i) {
        aK0[i]     = (_Float16)krow[hi * 8      + i];
        aK0[8 + i] = (_Float16)krow[hi * 8 + 16 + i];
        aK1[i]     = (_Float16)krow[hi * 8 + 32 + i];
        aK1[8 + i] = (_Float16)krow[hi * 8 + 48 + i];
    }
#pragma unroll
    for (int i = 0; i < 16; ++i) {
        bQ0[i] = (_Float16)qrow[hi * 8 + i + (i >= 8 ? 8 : 0)];
        bQ1[i] = (_Float16)qrow[hi * 8 + 32 + i + (i >= 8 ? 8 : 0)];
    }

    // S^T accumulated in f32 over the two 32-wide d-chunks.
    v8f st = {};
    st = __builtin_amdgcn_wmma_f32_16x16x32_f16(false, aK0, false, bQ0,
                                                (short)0, st, false, false);
    st = __builtin_amdgcn_wmma_f32_16x16x32_f16(false, aK1, false, bQ1,
                                                (short)0, st, false, false);

    // C layout: lane l holds S^T[n][m] = S[m][n] with m=l&15, n=hi*8+v.
    // => each valid lane (hi == (l&15)>>3, i.e. lanes 0-7 and 24-31) owns one
    // full softmax row in its 8 registers. Softmax is purely per-lane.
    float mx = st[0];
#pragma unroll
    for (int i = 1; i < 8; ++i) mx = fmaxf(mx, st[i]);
    float p[8];
    float sum = 0.f;
#pragma unroll
    for (int i = 0; i < 8; ++i) {
        p[i] = __expf((st[i] - mx) * SCALE);   // softmax(SCALE*s): exp<=1, safe everywhere
        sum += p[i];
    }
    const float inv = 1.0f / sum;

    // ---- B' = attn^T (32x16 f16, K=n padded 16..31 with zeros) ----
    // Row m of attn lives in lane m (m<8, K=0..7 in halves 0..7) or lane m+16
    // (m>=8, K=8..15 in halves 0..7); halves 8..15 are the zero K-padding.
    v16h bp = {};
#pragma unroll
    for (int i = 0; i < 8; ++i) {
        float an  = p[i] * inv;
        float rem = __shfl_xor(an, 16, 32);
        bp[i]     = (lane < 8 || lane >= 24) ? (_Float16)an  : (_Float16)0.0f;
        bp[8 + i] = (_Float16)0.0f;
    }

    // ---- O^T = V^T * attn^T, four 16-wide d-tiles ----
    // A' layout: lane l is row d = (l&15)+16*t; halves i<8 -> n = hi*8+i,
    // halves 8..15 -> n>=16 (zero pad). Loads of V columns are coalesced
    // across lanes (fixed n, consecutive d).
    const float* vcol = vg + base + r;
#pragma unroll
    for (int t = 0; t < 4; ++t) {
        v16h av = {};
#pragma unroll
        for (int i = 0; i < 8; ++i)
            av[i] = (_Float16)vcol[(size_t)(hi * 8 + i) * DIM + t * 16];

        v8f o = {};
        o = __builtin_amdgcn_wmma_f32_16x16x32_f16(false, av, false, bp,
                                                   (short)0, o, false, false);

        // O^T C layout: lane l holds O[m][d] with m=l&15, d=t*16+hi*8+v:
        // 8 contiguous floats per lane -> two b128 stores.
        float* orow = og + base + (size_t)r * DIM + t * 16 + hi * 8;
        f4 lo = { o[0], o[1], o[2], o[3] };
        f4 hx = { o[4], o[5], o[6], o[7] };
        *(f4*)(orow)     = lo;
        *(f4*)(orow + 4) = hx;
    }
}

extern "C" void kernel_launch(void* const* d_in, const int* in_sizes, int n_in,
                              void* d_out, int out_size, void* d_ws, size_t ws_size,
                              hipStream_t stream)
{
    const float* q = (const float*)d_in[0];
    const float* k = (const float*)d_in[1];
    const float* v = (const float*)d_in[2];
    float* out = (float*)d_out;

    const int npos   = in_sizes[0] / (HEADS * DIM);  // B*S = 16384
    const int blocks = (npos + 7) / 8;               // 8 waves (positions) per block

    hipLaunchKernelGGL(hilbert_head_attn_kernel, dim3(blocks), dim3(256), 0, stream,
                       q, k, v, out, npos);
}
